// GoogLeNet_5935644803282
// MI455X (gfx1250) — 8-bit matrix operand order repaired and checked by reading, compiled, not run on hardware
//
#include <hip/hip_runtime.h>
#include <stdint.h>

typedef __attribute__((ext_vector_type(8))) int v8i;

#define NB     32
#define CIN    64
#define COUT   192
#define HW     56
#define TW     12
#define VDIM   7
#define KCH    3
#define OTILES (COUT/16)
#define MROWS  (NB*HW*TW)
#define MTILES (MROWS/16)

__device__ __constant__ float BTf[7][7] = {
  { 1.0f, -1.0f/3.0f, -5.0f/4.0f,  5.0f/12.0f,  1.0f/4.0f, -1.0f/12.0f, 0.0f },
  { 0.0f,  1.0f,       2.0f/3.0f, -7.0f/12.0f, -1.0f/6.0f,  1.0f/12.0f, 0.0f },
  { 0.0f, -1.0f/2.0f,  2.0f/3.0f, -1.0f/24.0f, -1.0f/6.0f,  1.0f/24.0f, 0.0f },
  { 0.0f, -1.0f/4.0f, -1.0f/24.0f, 7.0f/24.0f,  1.0f/24.0f,-1.0f/24.0f, 0.0f },
  { 0.0f,  1.0f/20.0f,-1.0f/24.0f,-1.0f/24.0f,  1.0f/24.0f,-1.0f/120.0f,0.0f },
  { 0.0f,  1.0f/30.0f, 0.0f,      -1.0f/24.0f,  0.0f,       1.0f/120.0f,0.0f },
  { 0.0f,-12.0f,       4.0f,      15.0f,       -5.0f,      -3.0f,       1.0f }
};
__device__ __constant__ float ATf[5][7] = {
  {1,1, 1, 1, 1, 1,0},
  {0,1,-1, 2,-2, 3,0},
  {0,1, 1, 4, 4, 9,0},
  {0,1,-1, 8,-8,27,0},
  {0,1, 1,16,16,81,1}
};
__device__ __constant__ float Gf[7][3] = {
  {1,0,0},{1,1,1},{1,-1,1},{1,2,4},{1,-2,4},{1,3,9},{0,0,1}
};

__device__ __forceinline__ int a8_k(int reg, int lane, int b) {
  return ((reg >> 2) << 5) + (((reg >> 1) & 1) << 4) + ((lane >> 4) << 3) + ((reg & 1) << 2) + b;
}
__device__ __forceinline__ int b8_k(int reg, int lane) {
  return ((reg >> 2) << 5) + ((lane >> 4) << 4) + (lane & 15);
}
__device__ __forceinline__ int b8_n(int reg, int b) { return ((reg & 3) << 2) + b; }

__global__ void pack_B(const float* __restrict__ w,
                       uint32_t* __restrict__ bhi, uint32_t* __restrict__ blo) {
  int idx = blockIdx.x * blockDim.x + threadIdx.x;
  const int total = VDIM * KCH * OTILES * 32 * 8;
  if (idx >= total) return;
  int reg  = idx & 7;
  int lane = (idx >> 3) & 31;
  int rest = idx >> 8;
  int ot = rest % OTILES; rest /= OTILES;
  int kc = rest % KCH;    int v = rest / KCH;
  uint32_t hi = 0, lo = 0;
  for (int b = 0; b < 4; ++b) {
    int kl = a8_k(reg, lane, b);
    int kg = kc * 64 + kl;
    int c = kg / 3, r = kg % 3;
    int o = ot * 16 + (lane & 15);
    float s = 0.0f;
    for (int j = 0; j < 3; ++j)
      s = fmaf(Gf[v][j], w[((o * CIN + c) * 3 + r) * 3 + j], s);
    float rs = rintf(s);
    rs = fminf(fmaxf(rs, -32768.0f), 32767.0f);
    int wi = (int)rs;
    hi |= ((uint32_t)((wi >> 8) & 0xff)) << (8 * b);
    lo |= ((uint32_t)(wi & 0xff)) << (8 * b);
  }
  bhi[idx] = hi; blo[idx] = lo;
}

__global__ void pack_A(const float* __restrict__ x,
                       uint32_t* __restrict__ ahi, uint32_t* __restrict__ alo) {
  int idx = blockIdx.x * blockDim.x + threadIdx.x;
  const int total = VDIM * MTILES * KCH * 32 * 8;
  if (idx >= total) return;
  int reg  = idx & 7;
  int lane = (idx >> 3) & 31;
  int rest = idx >> 8;
  int kc = rest % KCH; rest /= KCH;
  int mt = rest % MTILES; int v = rest / MTILES;
  int mrow = mt * 16 + (lane & 15);
  int n = mrow / (HW * TW); int rem = mrow % (HW * TW);
  int k = rem / TW; int t = rem % TW;
  uint32_t hi = 0, lo = 0;
  for (int b = 0; b < 4; ++b) {
    int kl = a8_k(reg, lane, b);
    int kg = kc * 64 + kl;
    int c = kg / 3, r = kg % 3;
    int h = k + r;
    float s = 0.0f;
    if (h >= 1 && h <= HW) {
      const float* xb = x + ((long)(n * CIN + c) * HW + (h - 1)) * HW;
      for (int wi = 0; wi < 7; ++wi) {
        int col = t * 5 + wi - 1;
        if (col >= 0 && col < HW) s = fmaf(BTf[v][wi], xb[col], s);
      }
    }
    float rs = rintf(s);
    rs = fminf(fmaxf(rs, -32768.0f), 32767.0f);
    int e = (int)rs;
    hi |= ((uint32_t)((e >> 8) & 0xff)) << (8 * b);
    lo |= ((uint32_t)(e & 0xff)) << (8 * b);
  }
  ahi[idx] = hi; alo[idx] = lo;
}

__global__ __launch_bounds__(192) void winograd_main(
    const uint32_t* __restrict__ ahi, const uint32_t* __restrict__ alo,
    const uint32_t* __restrict__ bhi, const uint32_t* __restrict__ blo,
    const float* __restrict__ alpha, const float* __restrict__ beta,
    const float* __restrict__ sfp, const float* __restrict__ sop,
    float* __restrict__ out) {
  __shared__ uint32_t lA[2][256];

  int mt   = blockIdx.x;
  int og   = blockIdx.y;
  int tid  = threadIdx.x;
  int wave = tid >> 5;
  int lane = tid & 31;
  int ot   = og * 6 + wave;

  uint32_t ldsbase = (uint32_t)(uintptr_t)(void*)&lA[0][0];

  float yf[5][8];
  for (int u = 0; u < 5; ++u)
    for (int j = 0; j < 8; ++j) yf[u][j] = 0.0f;

  for (int v = 0; v < VDIM; ++v) {
    v8i acc_hh = {0,0,0,0,0,0,0,0};
    v8i acc_hl = {0,0,0,0,0,0,0,0};
    v8i acc_lh = {0,0,0,0,0,0,0,0};
    v8i acc_ll = {0,0,0,0,0,0,0,0};

    for (int kc = 0; kc < KCH; ++kc) {
      long abase = (((long)v * MTILES + mt) * KCH + kc) * 256;
      __syncthreads();

      if (wave < 4) {
        int half = wave >> 1;
        int seg  = wave & 1;
        const uint32_t* src = (half ? alo : ahi) + abase + seg * 128 + lane * 4;
        uint32_t dst = ldsbase + (uint32_t)(half * 1024 + seg * 512 + lane * 16);
        asm volatile("global_load_async_to_lds_b128 %0, %1, off"
                     :: "v"(dst), "v"((uint64_t)(uintptr_t)src)
                     : "memory");
        asm volatile("s_wait_asynccnt 0" ::: "memory");
      }
      __syncthreads();

      if (!(v == VDIM - 1 && kc == KCH - 1)) {
        long anext = abase + 256;
        __builtin_prefetch(&ahi[anext], 0, 1);
        __builtin_prefetch(&alo[anext], 0, 1);
      }

      v8i Ah, Al, Bh, Bl;
      const uint32_t* ph = &lA[0][lane * 8];
      const uint32_t* pl = &lA[1][lane * 8];
      long bbase = ((((long)v * KCH + kc) * OTILES + ot) * 32 + lane) * 8;
      #pragma unroll
      for (int rg = 0; rg < 8; ++rg) {
        Ah[rg] = (int)ph[rg];
        Al[rg] = (int)pl[rg];
        Bh[rg] = (int)bhi[bbase + rg];
        Bl[rg] = (int)blo[bbase + rg];
      }
      acc_hh = __builtin_amdgcn_wmma_i32_16x16x64_iu8(true,  Ah, true,  Bh, acc_hh, false, false);
      acc_hl = __builtin_amdgcn_wmma_i32_16x16x64_iu8(true,  Ah, false, Bl, acc_hl, false, false);
      acc_lh = __builtin_amdgcn_wmma_i32_16x16x64_iu8(false, Al, true,  Bh, acc_lh, false, false);
      acc_ll = __builtin_amdgcn_wmma_i32_16x16x64_iu8(false, Al, false, Bl, acc_ll, false, false);
    }

    #pragma unroll
    for (int j = 0; j < 8; ++j) {
      int mv = ((acc_hh[j] << 8) + acc_hl[j] + acc_lh[j]);
      mv = (mv << 8) + acc_ll[j];
      float mvf = (float)mv;
      #pragma unroll
      for (int u = 0; u < 5; ++u) yf[u][j] = fmaf(ATf[u][v], mvf, yf[u][j]);
    }
  }

  int col = lane & 15;
  int o = ot * 16 + col;
  float al = alpha[o];
  float be = beta[o];
  float sf = sfp[0];
  float so = sop[0];
  float bterm = rintf(be * so);
  float inv = 1.0f / (120.0f * sf);

  for (int j = 0; j < 8; ++j) {
    int mrow = mt * 16 + j + ((lane >> 4) << 3);
    int n = mrow / (HW * TW); int rem = mrow % (HW * TW);
    int k = rem / TW; int t = rem % TW;
    for (int u = 0; u < 5; ++u) {
      int wcol = t * 5 + u;
      if (wcol < HW) {
        float yv = rintf(yf[u][j] * inv);
        yv = rintf(al * yv) + bterm;
        yv = rintf(yv);
        yv = fminf(fmaxf(yv, -128.0f), 127.0f);
        yv = fmaxf(yv, 0.0f);
        out[(((long)n * COUT + o) * HW + k) * HW + wcol] = yv;
      }
    }
  }
}

extern "C" void kernel_launch(void* const* d_in, const int* in_sizes, int n_in,
                              void* d_out, int out_size, void* d_ws, size_t ws_size,
                              hipStream_t stream) {
  const float* x     = (const float*)d_in[0];
  const float* w     = (const float*)d_in[1];
  const float* alpha = (const float*)d_in[2];
  const float* beta  = (const float*)d_in[3];
  const float* sf    = (const float*)d_in[4];
  const float* so    = (const float*)d_in[5];
  float* out = (float*)d_out;

  const size_t awords = (size_t)VDIM * MTILES * KCH * 256;
  const size_t bwords = (size_t)VDIM * KCH * OTILES * 256;
  uint32_t* ahi = (uint32_t*)d_ws;
  uint32_t* alo = ahi + awords;
  uint32_t* bhi = alo + awords;
  uint32_t* blo = bhi + bwords;

  pack_B<<<(int)((bwords + 255) / 256), 256, 0, stream>>>(w, bhi, blo);
  pack_A<<<(int)((awords + 255) / 256), 256, 0, stream>>>(x, ahi, alo);

  dim3 grid(MTILES, 2);
  winograd_main<<<grid, 192, 0, stream>>>(ahi, alo, bhi, blo, alpha, beta, sf, so, out);
}
